// AttentionBlock_1924145348714
// MI455X (gfx1250) — matrix operand order repaired and checked by reading, not compiled, not run on hardware
//
#include <hip/hip_runtime.h>
#include <math.h>

typedef __attribute__((ext_vector_type(16))) __bf16 v16bf;
typedef __attribute__((ext_vector_type(8)))  float  v8f;
typedef __attribute__((ext_vector_type(2)))  __bf16 bf16x2;

#define S_  2048
#define DM_ 1024
#define H_  16
#define HD_ 64

// ---- CDNA5 async memory->LDS helpers -------------------------------------
// Generic LDS pointers carry the LDS byte offset in their low 32 bits
// (flat aperture mapping, ISA 10.2); generic global pointers are the 64-bit
// global address.  GLOBAL_LOAD_ASYNC_TO_LDS_B128 is tracked by ASYNCcnt.
__device__ __forceinline__ void async_b128(void* lds, const void* g) {
  unsigned l = (unsigned)(unsigned long long)(size_t)lds;
  unsigned long long ga = (unsigned long long)(size_t)g;
  asm volatile("global_load_async_to_lds_b128 %0, %1, off"
               :: "v"(l), "v"(ga) : "memory");
}
__device__ __forceinline__ void wait_async0() {
  asm volatile("s_wait_asynccnt 0" ::: "memory");
}

// ---------------------------------------------------------------- converts
__global__ void cvt_f32_to_bf16(const float* __restrict__ src,
                                __bf16* __restrict__ dst, int n) {
  int i = blockIdx.x * blockDim.x + threadIdx.x;
  if (i < n) dst[i] = (__bf16)src[i];
}

// ---------------------------------------------------------------- GEMM
// C[row, col] = A[row, :] . W[col, :]  (A: [M][1024] bf16, W: [1024][1024] bf16)
// mode 0/1: Q/K -> +bias, store bf16 [h][s][d]
// mode 2:   V   -> store bf16 transposed [h][d][s]
// mode 3:   O   -> +residual(x), store f32 [s][dm]
__global__ __launch_bounds__(256)
void gemm_bf16(const __bf16* __restrict__ A, const __bf16* __restrict__ W,
               const float* __restrict__ bias, const float* __restrict__ resid,
               __bf16* __restrict__ out_bf, float* __restrict__ out_f32, int mode) {
  __shared__ __align__(16) __bf16 lds_a[2][128][32];
  __shared__ __align__(16) __bf16 lds_b[2][64][32];
  const int tid = threadIdx.x;
  const int wid = tid >> 5, lane = tid & 31;
  const int half = lane >> 4, l16 = lane & 15;
  const int wm = (wid & 3) * 32;          // wave M offset within 128
  const int wn = (wid >> 2) * 32;         // wave N offset within 64
  const int row0 = blockIdx.y * 128, col0 = blockIdx.x * 64;

  // per-thread staging coordinates
  const int av0 = tid * 2;                              // A: 2 uint4/thread
  const int ar0 = av0 >> 2, ac0 = (av0 & 3) * 8;
  const int ar1 = (av0 + 1) >> 2, ac1 = ((av0 + 1) & 3) * 8;
  const int br = tid >> 2, bc = (tid & 3) * 8;          // B: 1 uint4/thread

  v8f acc[4] = {};

  // prologue: async-stage k-chunk 0 into buffer 0
  async_b128(&lds_a[0][ar0][ac0], &A[(size_t)(row0 + ar0) * DM_ + ac0]);
  async_b128(&lds_a[0][ar1][ac1], &A[(size_t)(row0 + ar1) * DM_ + ac1]);
  async_b128(&lds_b[0][br][bc],   &W[(size_t)(col0 + br) * DM_ + bc]);
  wait_async0();
  __syncthreads();

  int buf = 0;
  for (int kc = 0; kc < DM_; kc += 32) {
    if (kc + 32 < DM_) {                 // async-prefetch next chunk
      int kn = kc + 32;
      async_b128(&lds_a[buf ^ 1][ar0][ac0], &A[(size_t)(row0 + ar0) * DM_ + kn + ac0]);
      async_b128(&lds_a[buf ^ 1][ar1][ac1], &A[(size_t)(row0 + ar1) * DM_ + kn + ac1]);
      async_b128(&lds_b[buf ^ 1][br][bc],   &W[(size_t)(col0 + br) * DM_ + kn + bc]);
    }

    v16bf af[2], bfr[2];
    #pragma unroll
    for (int mt = 0; mt < 2; ++mt)
      #pragma unroll
      for (int r = 0; r < 8; ++r) {
        int k = (r < 4 ? 2 * r : 8 + 2 * r) + 8 * half;
        bf16x2 p = *(const bf16x2*)&lds_a[buf][wm + mt * 16 + l16][k];
        af[mt][2 * r] = p.x; af[mt][2 * r + 1] = p.y;
      }
    #pragma unroll
    for (int nt = 0; nt < 2; ++nt)
      #pragma unroll
      for (int r = 0; r < 8; ++r) {
        int k = (r < 4 ? 2 * r : 8 + 2 * r) + 8 * half;
        bf16x2 p = *(const bf16x2*)&lds_b[buf][wn + nt * 16 + l16][k];
        bfr[nt][2 * r] = p.x; bfr[nt][2 * r + 1] = p.y;
      }
    #pragma unroll
    for (int mt = 0; mt < 2; ++mt)
      #pragma unroll
      for (int nt = 0; nt < 2; ++nt)
        acc[mt * 2 + nt] = __builtin_amdgcn_wmma_f32_16x16x32_bf16(
            false, af[mt], false, bfr[nt], (short)0, acc[mt * 2 + nt], false, false);

    wait_async0();
    __syncthreads();
    buf ^= 1;
  }

  #pragma unroll
  for (int mt = 0; mt < 2; ++mt)
    #pragma unroll
    for (int nt = 0; nt < 2; ++nt)
      #pragma unroll
      for (int r = 0; r < 8; ++r) {
        int row = row0 + wm + mt * 16 + 8 * half + r;
        int col = col0 + wn + nt * 16 + l16;
        float v = acc[mt * 2 + nt][r];
        if (mode <= 1) {
          v += bias[(size_t)row * DM_ + col];
          int h = col >> 6, d = col & 63;
          out_bf[((size_t)h * S_ + row) * HD_ + d] = (__bf16)v;
        } else if (mode == 2) {
          int h = col >> 6, d = col & 63;
          out_bf[((size_t)h * HD_ + d) * S_ + row] = (__bf16)v;
        } else {
          out_f32[(size_t)row * DM_ + col] = v + resid[(size_t)row * DM_ + col];
        }
      }
}

// ---------------------------------------------------------------- attention
__global__ __launch_bounds__(256)
void attn_kernel(const __bf16* __restrict__ Q, const __bf16* __restrict__ K,
                 const __bf16* __restrict__ VT,
                 const unsigned char* __restrict__ mask,
                 __bf16* __restrict__ Obf) {
  __shared__ __align__(16) __bf16 lds_k[2][32][64];    // [key][d]
  __shared__ __align__(16) __bf16 lds_vt[2][64][32];   // [d][key]
  __shared__ __align__(16) __bf16 lds_p[8][16][32];    // per-wave P scratch
  const int tid = threadIdx.x, wid = tid >> 5, lane = tid & 31;
  const int half = lane >> 4, l16 = lane & 15;
  const int head = blockIdx.y;
  const int q0 = blockIdx.x * 128 + wid * 16;

  const int kr = tid >> 3, kc_ = (tid & 7) * 8;        // K stage coords
  const int vd = tid >> 2, vc = (tid & 3) * 8;         // VT stage coords

  // stage key-block 0 asynchronously; overlap with Q-fragment loads
  async_b128(&lds_k[0][kr][kc_], &K[((size_t)head * S_ + kr) * HD_ + kc_]);
  async_b128(&lds_vt[0][vd][vc], &VT[((size_t)head * HD_ + vd) * S_ + vc]);

  v16bf aq[2];                                         // Q fragments (hd = 2x32)
  #pragma unroll
  for (int c = 0; c < 2; ++c)
    #pragma unroll
    for (int r = 0; r < 8; ++r) {
      int k = c * 32 + (r < 4 ? 2 * r : 8 + 2 * r) + 8 * half;
      bf16x2 p = *(const bf16x2*)&Q[((size_t)head * S_ + q0 + l16) * HD_ + k];
      aq[c][2 * r] = p.x; aq[c][2 * r + 1] = p.y;
    }

  v8f acc[4] = {};
  float mrow[8], lrow[8];
  #pragma unroll
  for (int r = 0; r < 8; ++r) { mrow[r] = -1e30f; lrow[r] = 0.f; }

  wait_async0();
  __syncthreads();

  int buf = 0;
  for (int kb = 0; kb < S_; kb += 32) {
    if (kb + 32 < S_) {                  // async-prefetch next key block
      async_b128(&lds_k[buf ^ 1][kr][kc_],
                 &K[((size_t)head * S_ + kb + 32 + kr) * HD_ + kc_]);
      async_b128(&lds_vt[buf ^ 1][vd][vc],
                 &VT[((size_t)head * HD_ + vd) * S_ + kb + 32 + vc]);
    }

    float p[2][8];
    #pragma unroll
    for (int t = 0; t < 2; ++t) {                      // two 16x16 logit tiles
      v8f lg = {};
      #pragma unroll
      for (int c = 0; c < 2; ++c) {
        v16bf bk;
        #pragma unroll
        for (int r = 0; r < 8; ++r) {
          int k = c * 32 + (r < 4 ? 2 * r : 8 + 2 * r) + 8 * half;
          bf16x2 q2 = *(const bf16x2*)&lds_k[buf][t * 16 + l16][k];
          bk[2 * r] = q2.x; bk[2 * r + 1] = q2.y;
        }
        lg = __builtin_amdgcn_wmma_f32_16x16x32_bf16(
            false, aq[c], false, bk, (short)0, lg, false, false);
      }
      #pragma unroll
      for (int r = 0; r < 8; ++r) {
        int rowq = q0 + 8 * half + r;
        int kcol = kb + t * 16 + l16;
        float lv = lg[r] * 0.125f;                     // 1/sqrt(64)
        p[t][r] = mask[(size_t)rowq * S_ + kcol] ? lv : -1e30f;
      }
    }
    float scale[8];
    #pragma unroll
    for (int r = 0; r < 8; ++r) {                      // online softmax update
      float mx = fmaxf(p[0][r], p[1][r]);
      for (int m = 8; m >= 1; m >>= 1) mx = fmaxf(mx, __shfl_xor(mx, m, 32));
      float mnew = fmaxf(mrow[r], mx);
      scale[r] = __expf(mrow[r] - mnew);
      p[0][r] = __expf(p[0][r] - mnew);
      p[1][r] = __expf(p[1][r] - mnew);
      float s = p[0][r] + p[1][r];
      for (int m = 8; m >= 1; m >>= 1) s += __shfl_xor(s, m, 32);
      lrow[r] = lrow[r] * scale[r] + s;
      mrow[r] = mnew;
    }
    #pragma unroll
    for (int nt = 0; nt < 4; ++nt)
      #pragma unroll
      for (int r = 0; r < 8; ++r) acc[nt][r] *= scale[r];

    #pragma unroll
    for (int t = 0; t < 2; ++t)                        // C-layout -> LDS
      #pragma unroll
      for (int r = 0; r < 8; ++r)
        lds_p[wid][8 * half + r][t * 16 + l16] = (__bf16)p[t][r];
    asm volatile("s_wait_dscnt 0" ::: "memory");       // wave-local LDS RAW

    v16bf ap;                                          // reload in A layout
    #pragma unroll
    for (int r = 0; r < 8; ++r) {
      int k = (r < 4 ? 2 * r : 8 + 2 * r) + 8 * half;
      bf16x2 p2 = *(const bf16x2*)&lds_p[wid][l16][k];
      ap[2 * r] = p2.x; ap[2 * r + 1] = p2.y;
    }
    #pragma unroll
    for (int nt = 0; nt < 4; ++nt) {                   // acc += P @ V
      v16bf bv;
      #pragma unroll
      for (int r = 0; r < 8; ++r) {
        int k = (r < 4 ? 2 * r : 8 + 2 * r) + 8 * half;
        bf16x2 v2 = *(const bf16x2*)&lds_vt[buf][nt * 16 + l16][k];
        bv[2 * r] = v2.x; bv[2 * r + 1] = v2.y;
      }
      acc[nt] = __builtin_amdgcn_wmma_f32_16x16x32_bf16(
          false, ap, false, bv, (short)0, acc[nt], false, false);
    }

    wait_async0();
    __syncthreads();
    buf ^= 1;
  }

  #pragma unroll
  for (int nt = 0; nt < 4; ++nt)
    #pragma unroll
    for (int r = 0; r < 8; ++r) {
      int row = q0 + 8 * half + r;
      int col = head * HD_ + nt * 16 + l16;
      float inv = lrow[r] > 0.f ? 1.f / lrow[r] : 0.f;
      Obf[(size_t)row * DM_ + col] = (__bf16)(acc[nt][r] * inv);
    }
}

// ---------------------------------------------------------------- layernorm
__global__ __launch_bounds__(256)
void ln_kernel(const float* __restrict__ res, const float* __restrict__ gamma,
               const float* __restrict__ beta, float* __restrict__ out) {
  __shared__ float red[256];
  const int row = blockIdx.x, tid = threadIdx.x;
  float v[4], s = 0.f;
  #pragma unroll
  for (int i = 0; i < 4; ++i) {
    v[i] = res[(size_t)row * DM_ + tid + i * 256];
    s += v[i];
  }
  red[tid] = s; __syncthreads();
  for (int o = 128; o > 0; o >>= 1) {
    if (tid < o) red[tid] += red[tid + o];
    __syncthreads();
  }
  float mu = red[0] / DM_; __syncthreads();
  float s2 = 0.f;
  #pragma unroll
  for (int i = 0; i < 4; ++i) { float d = v[i] - mu; s2 += d * d; }
  red[tid] = s2; __syncthreads();
  for (int o = 128; o > 0; o >>= 1) {
    if (tid < o) red[tid] += red[tid + o];
    __syncthreads();
  }
  float rstd = rsqrtf(red[0] / DM_ + 1e-5f);
  #pragma unroll
  for (int i = 0; i < 4; ++i) {
    int c = tid + i * 256;
    out[(size_t)row * DM_ + c] = (v[i] - mu) * rstd * gamma[c] + beta[c];
  }
}

// ---------------------------------------------------------------- launch
extern "C" void kernel_launch(void* const* d_in, const int* in_sizes, int n_in,
                              void* d_out, int out_size, void* d_ws,
                              size_t ws_size, hipStream_t stream) {
  const float* x     = (const float*)d_in[0];
  const float* bias  = (const float*)d_in[1];   // static_bias [s][h][hd]=[s][dm]
  const float* Wq    = (const float*)d_in[2];
  const float* Wk    = (const float*)d_in[3];
  const float* Wv    = (const float*)d_in[4];
  const float* Wo    = (const float*)d_in[5];
  const float* gamma = (const float*)d_in[6];
  const float* beta  = (const float*)d_in[7];
  const unsigned char* mask = (const unsigned char*)d_in[8];
  float* out = (float*)d_out;

  char* ws = (char*)d_ws;
  __bf16* xb    = (__bf16*)(ws + (size_t)(0)  * (1 << 20));
  __bf16* wqb   = (__bf16*)(ws + (size_t)(4)  * (1 << 20));
  __bf16* wkb   = (__bf16*)(ws + (size_t)(6)  * (1 << 20));
  __bf16* wvb   = (__bf16*)(ws + (size_t)(8)  * (1 << 20));
  __bf16* wob   = (__bf16*)(ws + (size_t)(10) * (1 << 20));
  __bf16* qh    = (__bf16*)(ws + (size_t)(12) * (1 << 20));  // [h][s][d]
  __bf16* kh    = (__bf16*)(ws + (size_t)(16) * (1 << 20));  // [h][s][d]
  __bf16* vt    = (__bf16*)(ws + (size_t)(20) * (1 << 20));  // [h][d][s]
  __bf16* attnb = (__bf16*)(ws + (size_t)(24) * (1 << 20));  // [s][dm]
  float*  res   = (float*) (ws + (size_t)(28) * (1 << 20));  // [s][dm]

  const int nx = S_ * DM_, nw = DM_ * DM_;
  cvt_f32_to_bf16<<<(nx + 255) / 256, 256, 0, stream>>>(x,  xb,  nx);
  cvt_f32_to_bf16<<<(nw + 255) / 256, 256, 0, stream>>>(Wq, wqb, nw);
  cvt_f32_to_bf16<<<(nw + 255) / 256, 256, 0, stream>>>(Wk, wkb, nw);
  cvt_f32_to_bf16<<<(nw + 255) / 256, 256, 0, stream>>>(Wv, wvb, nw);
  cvt_f32_to_bf16<<<(nw + 255) / 256, 256, 0, stream>>>(Wo, wob, nw);

  dim3 gg(DM_ / 64, S_ / 128);
  gemm_bf16<<<gg, 256, 0, stream>>>(xb, wqb, bias, nullptr, qh, nullptr, 0);
  gemm_bf16<<<gg, 256, 0, stream>>>(xb, wkb, bias, nullptr, kh, nullptr, 1);
  gemm_bf16<<<gg, 256, 0, stream>>>(xb, wvb, nullptr, nullptr, vt, nullptr, 2);

  dim3 ga(S_ / 128, H_);
  attn_kernel<<<ga, 256, 0, stream>>>(qh, kh, vt, mask, attnb);

  gemm_bf16<<<gg, 256, 0, stream>>>(attnb, wob, nullptr, x, nullptr, res, 3);
  ln_kernel<<<S_, 256, 0, stream>>>(res, gamma, beta, out);
}
